// MultiheadedSelfAttention_90142773608709
// MI455X (gfx1250) — matrix operand order repaired and checked by reading, not compiled, not run on hardware
//
#include <hip/hip_runtime.h>
#include <hip/hip_bf16.h>

// ---------------------------------------------------------------------------
// Multi-head self-attention for MI455X (gfx1250, wave32, WMMA).
// B=8, S=1024, D=1024, H=16, HD=64.
// All GEMMs use v_wmma_f32_16x16x32_bf16 (fp32 rounded to bf16, f32 accum).
// attn_prob (512MB f32) dominates HBM traffic: written once (scores),
// normalized in place (softmax), re-read once (PV).
// LDS layouts give 2x ds_load_b128 per WMMA fragment; the scores kernel
// stages its (pure-copy) tiles with GLOBAL_LOAD_ASYNC_TO_LDS_B128.
// ---------------------------------------------------------------------------

#define BQ 8
#define SQ 1024
#define DQ 1024
#define HQ 16
#define HDQ 64
#define MQ (BQ * SQ)                     // 8192 rows of x
#define OUT_ELEMS (BQ * SQ * DQ)         // 8,388,608 floats (out), attn follows

typedef __attribute__((ext_vector_type(16))) __bf16 v16bf;
typedef __attribute__((ext_vector_type(8)))  float  v8f;

// float -> bf16 bits, round to nearest even
__device__ __forceinline__ unsigned int f32_to_bf16(float f) {
  unsigned int u = __float_as_uint(f);
  u += 0x7FFFu + ((u >> 16) & 1u);
  return u >> 16;
}
__device__ __forceinline__ unsigned int pack2_bf16(float lo, float hi) {
  return f32_to_bf16(lo) | (f32_to_bf16(hi) << 16);
}

// Async 16B global->LDS copy (ASYNCcnt-tracked; per-lane addresses).
// Generic pointers to __shared__ carry the LDS byte offset in their low 32
// bits on AMDGPU, which is what the VDST operand of the async op wants.
__device__ __forceinline__ void async_copy_b128(const void* gptr, void* lptr) {
  const unsigned lds = (unsigned)(unsigned long long)lptr;
  asm volatile("global_load_async_to_lds_b128 %0, %1, off"
               :
               : "v"(lds), "v"(gptr)
               : "memory");
}
__device__ __forceinline__ void wait_asynccnt0() {
  asm volatile("s_wait_asynccnt 0" ::: "memory");
}

// -------- WMMA fragment loaders (ISA 7.12.2 layouts, wave32) ---------------
// A (16x32 bf16) from row-major LDS tile [16][lda]:
// lane L: row m=L&15, group g=L>>4. VGPR r<4: K=2r+8g ; r>=4: K=16+2(r-4)+8g.
// -> two aligned 16B runs at (m*lda + 8g) and (m*lda + 16 + 8g).
__device__ __forceinline__ v16bf load_a_frag(const unsigned short* a, int lda, int lane) {
  const int m = lane & 15;
  const int g = lane >> 4;
  const unsigned short* p = a + m * lda + 8 * g;
  union { uint4 q[2]; v16bf v; } u;
  u.q[0] = *(const uint4*)(p);
  u.q[1] = *(const uint4*)(p + 16);
  return u.v;
}

// B (32x16 bf16) from column-major LDS tile Bt[n][k] with stride ldk:
// lane L: col n=L&15, kb0 = 8*(L>>4); two 16B runs at K=kb0 and K=16+kb0.
__device__ __forceinline__ v16bf load_b_frag_t(const unsigned short* bt, int ldk, int lane) {
  const int n   = lane & 15;
  const int kb0 = (lane >> 4) * 8;
  const unsigned short* p = bt + n * ldk + kb0;
  union { uint4 q[2]; v16bf v; } u;
  u.q[0] = *(const uint4*)(p);
  u.q[1] = *(const uint4*)(p + 16);
  return u.v;
}

// ---------------------------------------------------------------------------
// Kernel 1: QKV projection. grid=(M/64, D/64, 3), block=128 (4 waves).
// ---------------------------------------------------------------------------
__global__ __launch_bounds__(128) void qkv_gemm_kernel(
    const float* __restrict__ x,
    const float* __restrict__ Wq, const float* __restrict__ bq,
    const float* __restrict__ Wk, const float* __restrict__ bk,
    const float* __restrict__ Wv, const float* __restrict__ bv,
    unsigned short* __restrict__ qbuf, unsigned short* __restrict__ kbuf,
    unsigned short* __restrict__ vbuf) {
  const float* W; const float* bias; unsigned short* out;
  if (blockIdx.z == 0)      { W = Wq; bias = bq; out = qbuf; }
  else if (blockIdx.z == 1) { W = Wk; bias = bk; out = kbuf; }
  else                      { W = Wv; bias = bv; out = vbuf; }

  const int m0 = blockIdx.x * 64;
  const int n0 = blockIdx.y * 64;
  const int t    = threadIdx.x;
  const int wave = t >> 5;
  const int lane = t & 31;

  __shared__ unsigned short sA [64 * 32];   // x tile  row-major [m][k]
  __shared__ unsigned short sBt[64 * 32];   // W tile  col-major [n][k]

  v8f acc[4] = {};

  for (int k0 = 0; k0 < DQ; k0 += 32) {
    if (k0 + 32 < DQ) {   // prefetch next K tile (global_prefetch_b8)
      __builtin_prefetch(x + (size_t)(m0 + (t >> 1)) * DQ + k0 + 32, 0, 1);
      __builtin_prefetch(W + (size_t)(k0 + 32 + (t >> 2)) * DQ + n0, 0, 1);
    }
    // stage A: 64x32 f32 -> bf16, float4 reads, packed uint2 LDS stores
#pragma unroll
    for (int j = 0; j < 4; ++j) {
      const int idx = t + j * 128;         // vec4 index, 8 per row
      const int r = idx >> 3;
      const int c = (idx & 7) * 4;
      const float4 f = *(const float4*)(x + (size_t)(m0 + r) * DQ + k0 + c);
      uint2 pk;
      pk.x = pack2_bf16(f.x, f.y);
      pk.y = pack2_bf16(f.z, f.w);
      *(uint2*)(sA + r * 32 + c) = pk;
    }
    // stage B transposed: coalesced float4 reads, scalar transposed stores
#pragma unroll
    for (int j = 0; j < 4; ++j) {
      const int idx = t + j * 128;         // vec4 index, 16 per row
      const int r = idx >> 4;              // k within tile
      const int c = (idx & 15) * 4;        // n within tile
      const float4 f = *(const float4*)(W + (size_t)(k0 + r) * DQ + n0 + c);
      sBt[(c + 0) * 32 + r] = (unsigned short)f32_to_bf16(f.x);
      sBt[(c + 1) * 32 + r] = (unsigned short)f32_to_bf16(f.y);
      sBt[(c + 2) * 32 + r] = (unsigned short)f32_to_bf16(f.z);
      sBt[(c + 3) * 32 + r] = (unsigned short)f32_to_bf16(f.w);
    }
    __syncthreads();

    const v16bf a = load_a_frag(sA + wave * 16 * 32, 32, lane);
#pragma unroll
    for (int j = 0; j < 4; ++j) {
      const v16bf b = load_b_frag_t(sBt + j * 16 * 32, 32, lane);
      acc[j] = __builtin_amdgcn_wmma_f32_16x16x32_bf16(
          false, a, false, b, (short)0, acc[j], false, false);
    }
    __syncthreads();
  }

  // Epilogue: + bias, store bf16. C layout: lane -> col N=lane&15,
  // VGPR v -> row M = v + 8*(lane>>4).
  const int ncl = lane & 15;
  const int g   = lane >> 4;
#pragma unroll
  for (int j = 0; j < 4; ++j) {
    const int n   = n0 + j * 16 + ncl;
    const float bb = bias[n];
#pragma unroll
    for (int v = 0; v < 8; ++v) {
      const int m = m0 + wave * 16 + v + 8 * g;
      out[(size_t)m * DQ + n] = (unsigned short)f32_to_bf16(acc[j][v] + bb);
    }
  }
}

// ---------------------------------------------------------------------------
// Kernel 2: scores = Q K^T / sqrt(D).  grid=(S/64, S/64, B*H), block=128.
// B = K^T, so the column-major B tile Bt[n][d] is just K row-major: the
// staging is a pure bf16 copy -> use async global->LDS b128 copies.
// ---------------------------------------------------------------------------
__global__ __launch_bounds__(128) void scores_kernel(
    const unsigned short* __restrict__ qbuf,
    const unsigned short* __restrict__ kbuf,
    float* __restrict__ attn) {
  const int z = blockIdx.z;          // b*H + h
  const int b = z >> 4;
  const int h = z & 15;
  const int q0 = blockIdx.x * 64;
  const int k0 = blockIdx.y * 64;
  const int t    = threadIdx.x;
  const int wave = t >> 5;
  const int lane = t & 31;
  const size_t rowb = (size_t)b * SQ;

  __shared__ unsigned short sQ[64 * 64];   // [q row][d]
  __shared__ unsigned short sK[64 * 64];   // [k row][d]  == Bt

  // stage via async global->LDS copies (16B per lane per op)
#pragma unroll
  for (int j = 0; j < 4; ++j) {
    const int idx = t + j * 128;           // vec8 index, 8 per row
    const int r = idx >> 3;
    const int c = (idx & 7) * 8;
    async_copy_b128(qbuf + (rowb + q0 + r) * DQ + h * HDQ + c, sQ + r * 64 + c);
    async_copy_b128(kbuf + (rowb + k0 + r) * DQ + h * HDQ + c, sK + r * 64 + c);
  }
  wait_asynccnt0();
  __syncthreads();

  v8f acc[4] = {};
#pragma unroll
  for (int d0 = 0; d0 < HDQ; d0 += 32) {
    const v16bf a = load_a_frag(sQ + wave * 16 * 64 + d0, 64, lane);
#pragma unroll
    for (int j = 0; j < 4; ++j) {
      const v16bf bf = load_b_frag_t(sK + j * 16 * 64 + d0, 64, lane);
      acc[j] = __builtin_amdgcn_wmma_f32_16x16x32_bf16(
          false, a, false, bf, (short)0, acc[j], false, false);
    }
  }

  const float inv_scale = 1.0f / 32.0f;    // 1/sqrt(1024)
  float* base = attn + (size_t)z * SQ * SQ;
  const int ncl = lane & 15;
  const int g   = lane >> 4;
#pragma unroll
  for (int j = 0; j < 4; ++j) {
    const int n = k0 + j * 16 + ncl;
#pragma unroll
    for (int v = 0; v < 8; ++v) {
      const int m = q0 + wave * 16 + v + 8 * g;
      base[(size_t)m * SQ + n] = acc[j][v] * inv_scale;
    }
  }
}

// ---------------------------------------------------------------------------
// Kernel 3: row softmax in place. grid = B*H*S blocks of 128 threads.
// ---------------------------------------------------------------------------
__global__ __launch_bounds__(128) void softmax_rows_kernel(float* __restrict__ attn) {
  float* p = attn + (size_t)blockIdx.x * SQ;
  const int t = threadIdx.x;
  __shared__ float red[128];

  float v[8];
  float m = -3.402823466e38f;
#pragma unroll
  for (int j = 0; j < 8; ++j) { v[j] = p[t + 128 * j]; m = fmaxf(m, v[j]); }
  red[t] = m; __syncthreads();
  for (int s = 64; s > 0; s >>= 1) {
    if (t < s) red[t] = fmaxf(red[t], red[t + s]);
    __syncthreads();
  }
  m = red[0]; __syncthreads();

  float sum = 0.f;
#pragma unroll
  for (int j = 0; j < 8; ++j) { v[j] = __expf(v[j] - m); sum += v[j]; }
  red[t] = sum; __syncthreads();
  for (int s = 64; s > 0; s >>= 1) {
    if (t < s) red[t] += red[t + s];
    __syncthreads();
  }
  const float inv = 1.0f / red[0];
#pragma unroll
  for (int j = 0; j < 8; ++j) p[t + 128 * j] = v[j] * inv;
}

// ---------------------------------------------------------------------------
// Kernel 4: out = P @ V.  grid=(S/64, B*H), block=128. N = HD = 64.
// ---------------------------------------------------------------------------
__global__ __launch_bounds__(128) void pv_gemm_kernel(
    const float* __restrict__ attn,
    const unsigned short* __restrict__ vbuf,
    float* __restrict__ out) {
  const int z = blockIdx.y;
  const int b = z >> 4;
  const int h = z & 15;
  const int m0 = blockIdx.x * 64;
  const int t    = threadIdx.x;
  const int wave = t >> 5;
  const int lane = t & 31;

  __shared__ unsigned short sP [64 * 32];  // probs tile row-major [m][k]
  __shared__ unsigned short sVt[64 * 32];  // V tile col-major [n][k]

  v8f acc[4] = {};
  const float* pbase = attn + (size_t)z * SQ * SQ;

  for (int k0 = 0; k0 < SQ; k0 += 32) {
    if (k0 + 32 < SQ) {
      __builtin_prefetch(pbase + (size_t)(m0 + (t >> 1)) * SQ + k0 + 32, 0, 1);
    }
    // stage P: f32 -> bf16 packed
#pragma unroll
    for (int j = 0; j < 4; ++j) {
      const int idx = t + j * 128;
      const int r = idx >> 3;
      const int c = (idx & 7) * 4;
      const float4 f = *(const float4*)(pbase + (size_t)(m0 + r) * SQ + k0 + c);
      uint2 pk;
      pk.x = pack2_bf16(f.x, f.y);
      pk.y = pack2_bf16(f.z, f.w);
      *(uint2*)(sP + r * 32 + c) = pk;
    }
    // stage V transposed: coalesced uint2 (4 bf16) reads, scalar stores
#pragma unroll
    for (int j = 0; j < 4; ++j) {
      const int idx = t + j * 128;
      const int r = idx >> 4;              // k within tile
      const int c = (idx & 15) * 4;        // n within tile
      const uint2 w = *(const uint2*)(vbuf + ((size_t)b * SQ + k0 + r) * DQ + h * HDQ + c);
      sVt[(c + 0) * 32 + r] = (unsigned short)(w.x & 0xffffu);
      sVt[(c + 1) * 32 + r] = (unsigned short)(w.x >> 16);
      sVt[(c + 2) * 32 + r] = (unsigned short)(w.y & 0xffffu);
      sVt[(c + 3) * 32 + r] = (unsigned short)(w.y >> 16);
    }
    __syncthreads();

    const v16bf a = load_a_frag(sP + wave * 16 * 32, 32, lane);
#pragma unroll
    for (int j = 0; j < 4; ++j) {
      const v16bf bf = load_b_frag_t(sVt + j * 16 * 32, 32, lane);
      acc[j] = __builtin_amdgcn_wmma_f32_16x16x32_bf16(
          false, a, false, bf, (short)0, acc[j], false, false);
    }
    __syncthreads();
  }

  const int ncl = lane & 15;
  const int g   = lane >> 4;
#pragma unroll
  for (int j = 0; j < 4; ++j) {
    const int n = h * HDQ + j * 16 + ncl;
#pragma unroll
    for (int v = 0; v < 8; ++v) {
      const int s = m0 + wave * 16 + v + 8 * g;
      out[((size_t)b * SQ + s) * DQ + n] = acc[j][v];
    }
  }
}

// ---------------------------------------------------------------------------
extern "C" void kernel_launch(void* const* d_in, const int* in_sizes, int n_in,
                              void* d_out, int out_size, void* d_ws, size_t ws_size,
                              hipStream_t stream) {
  (void)in_sizes; (void)n_in; (void)out_size; (void)ws_size;
  const float* x  = (const float*)d_in[0];
  const float* Wq = (const float*)d_in[1];
  const float* bq = (const float*)d_in[2];
  const float* Wk = (const float*)d_in[3];
  const float* bk = (const float*)d_in[4];
  const float* Wv = (const float*)d_in[5];
  const float* bv = (const float*)d_in[6];

  float* out  = (float*)d_out;                 // [B,S,D]
  float* attn = out + (size_t)OUT_ELEMS;       // [B,H,S,S]

  // Workspace: Q/K/V in bf16, each M*D elements (16 MB), 48 MB total.
  unsigned short* qbuf = (unsigned short*)d_ws;
  unsigned short* kbuf = qbuf + (size_t)MQ * DQ;
  unsigned short* vbuf = kbuf + (size_t)MQ * DQ;

  dim3 g1(MQ / 64, DQ / 64, 3);
  qkv_gemm_kernel<<<g1, 128, 0, stream>>>(x, Wq, bq, Wk, bk, Wv, bv,
                                          qbuf, kbuf, vbuf);
  dim3 g2(SQ / 64, SQ / 64, BQ * HQ);
  scores_kernel<<<g2, 128, 0, stream>>>(qbuf, kbuf, attn);
  softmax_rows_kernel<<<BQ * HQ * SQ, 128, 0, stream>>>(attn);
  dim3 g4(SQ / 64, BQ * HQ);
  pv_gemm_kernel<<<g4, 128, 0, stream>>>(attn, vbuf, out);
}
